// MultiHeadAttention_76407468195912
// MI455X (gfx1250) — matrix operand order repaired and checked by reading, not compiled, not run on hardware
//
#include <hip/hip_runtime.h>
#include <hip/hip_bf16.h>

// ---------------------------------------------------------------------------
// MI455X (gfx1250) multi-head attention, bf16 WMMA with f32 accumulate.
// Matrix ops: v_wmma_f32_16x16x32_bf16 (wave32).
// bf16 tiles: global_load_async_to_lds_b128 (ASYNCcnt DMA), double-buffered.
// f32 tiles: register-pipelined load -> cvt_pk_bf16 -> ds_store, + prefetch.
// ---------------------------------------------------------------------------

typedef __attribute__((ext_vector_type(8)))  float  v8f;
typedef __attribute__((ext_vector_type(8)))  __bf16 v8bf;
typedef __attribute__((ext_vector_type(16))) __bf16 v16bf;

union BFrag { v16bf v; v8bf h[2]; };

constexpr int Bc = 4, Lc = 2048, Dc = 1024, Hc = 16, HDc = 64;
constexpr int BM = 128, BN = 64, BK = 32;      // block tile
constexpr int LDSS = BK + 8;                   // 40 bf16 = 80 B row stride, 16B aligned
constexpr int NT = 256;                        // 8 waves

// ---- f32 tile staging, two-phase for software pipelining ------------------
template <int ROWS>
__device__ __forceinline__ void load_f32(float4* regs, const float* src, long ld,
                                         int tid) {
#pragma unroll
  for (int i = 0; i < ROWS * 8 / NT; ++i) {
    const int idx = tid + i * NT;
    const int r = idx >> 3, c = (idx & 7) << 2;
    regs[i] = *(const float4*)(src + (long)r * ld + c);
  }
}
template <int ROWS>
__device__ __forceinline__ void store_bf16(__bf16* dst, const float4* regs, int tid) {
#pragma unroll
  for (int i = 0; i < ROWS * 8 / NT; ++i) {
    const int idx = tid + i * NT;
    const int r = idx >> 3, c = (idx & 7) << 2;
    __bf16* p = dst + r * LDSS + c;
    p[0] = (__bf16)regs[i].x; p[1] = (__bf16)regs[i].y;
    p[2] = (__bf16)regs[i].z; p[3] = (__bf16)regs[i].w;
  }
}
// one global_prefetch_b8 per tile row (cacheline covers the 128B K-chunk)
template <int ROWS>
__device__ __forceinline__ void prefetch_rows(const float* src, long ld, int tid) {
  if (tid < ROWS) __builtin_prefetch(src + (long)tid * ld, 0, 1);
}

// ---- bf16 tile staging via async Global->LDS DMA (ASYNCcnt) ---------------
// 16 bytes (8 bf16) per lane-op; LDS address = low 32 bits of flat address.
template <int ROWS>
__device__ __forceinline__ void stage_async_bf16(__bf16* dst, const __bf16* src,
                                                 long ld, int tid) {
#pragma unroll
  for (int i = 0; i < ROWS * 4 / NT; ++i) {
    const int idx = tid + i * NT;
    const int r = idx >> 2, c = (idx & 3) << 3;
    unsigned           lds = (unsigned)(uintptr_t)(dst + r * LDSS + c);
    unsigned long long ga  = (unsigned long long)(uintptr_t)(src + (long)r * ld + c);
    asm volatile("global_load_async_to_lds_b128 %0, %1, off"
                 :: "v"(lds), "v"(ga) : "memory");
  }
}
// async loads complete in issue order -> threshold waits retire older tiles
__device__ __forceinline__ void wait_async0() { asm volatile("s_wait_asynccnt 0x0" ::: "memory"); }
__device__ __forceinline__ void wait_async1() { asm volatile("s_wait_asynccnt 0x1" ::: "memory"); }
__device__ __forceinline__ void wait_async2() { asm volatile("s_wait_asynccnt 0x2" ::: "memory"); }
__device__ __forceinline__ void wait_async3() { asm volatile("s_wait_asynccnt 0x3" ::: "memory"); }

// ---- one BK=32 K-step: 2x2 WMMA tiles per wave ----------------------------
// A-fragment (16x32 bf16): lane<16 -> M=lane, elems {K0..7, K16..23}
//                          lane>=16 -> M=lane-16, elems {K8..15, K24..31}
// B-fragment (32x16 bf16): lane<16 -> N=lane, {K0..7, K16..23} ; lane>=16 -> N=lane-16, {K8..15, K24..31}
__device__ __forceinline__ void kblock_mma(const __bf16* As, const __bf16* Bs,
                                           int wm, int wn, int lane, v8f acc[2][2]) {
  const int half = lane >> 4;
  const int l15  = lane & 15;
  BFrag a[2], b[2];
#pragma unroll
  for (int i = 0; i < 2; ++i) {
    const __bf16* ap = As + (wm * 32 + i * 16 + l15) * LDSS + half * 8;
    a[i].h[0] = *(const v8bf*)(ap);
    a[i].h[1] = *(const v8bf*)(ap + 16);
  }
#pragma unroll
  for (int j = 0; j < 2; ++j) {
    const __bf16* bp = Bs + (wn * 32 + j * 16 + l15) * LDSS + half * 8;
    b[j].h[0] = *(const v8bf*)(bp);
    b[j].h[1] = *(const v8bf*)(bp + 16);
  }
#pragma unroll
  for (int i = 0; i < 2; ++i)
#pragma unroll
    for (int j = 0; j < 2; ++j)
      acc[i][j] = __builtin_amdgcn_wmma_f32_16x16x32_bf16(
          false, a[i].v, false, b[j].v, (short)0, acc[i][j], false, false);
}

// C/D f32 16x16 layout: elem r -> (M = r or r+8 per lane-half, N = lane&15)
template <typename F>
__device__ __forceinline__ void epilogue(const v8f acc[2][2], int wm, int wn,
                                         int lane, long row0, long col0, F f) {
  const int half = lane >> 4, l15 = lane & 15;
#pragma unroll
  for (int i = 0; i < 2; ++i)
#pragma unroll
    for (int j = 0; j < 2; ++j)
#pragma unroll
      for (int r = 0; r < 8; ++r) {
        const long m = row0 + wm * 32 + i * 16 + (half ? r + 8 : r);
        const long n = col0 + wn * 32 + j * 16 + l15;
        f(m, n, acc[i][j][r]);
      }
}

// ---------------------------------------------------------------------------
// Kernel 1: projection  dst = X @ W^T   (X:[M,D] f32, W:[N,D] f32)
// Register-pipelined staging: store tile k, issue loads for k+1, prefetch k+2.
// mode 0: dst bf16 [B,H,L,HD]   mode 1: dst bf16 [B,H,HD,L] (transposed V)
// ---------------------------------------------------------------------------
__global__ __launch_bounds__(NT) void proj_kernel(const float* __restrict__ X,
                                                  const float* __restrict__ W,
                                                  __bf16* __restrict__ dst,
                                                  int transpose_out) {
  __shared__ alignas(16) __bf16 As[BM * LDSS];
  __shared__ alignas(16) __bf16 Bs[BN * LDSS];
  const int tid = threadIdx.x, lane = tid & 31, wid = tid >> 5;
  const int wm = wid & 3, wn = wid >> 2;
  const long row0 = (long)blockIdx.x * BM;
  const long col0 = (long)blockIdx.y * BN;
  const float* Ab = X + row0 * Dc;
  const float* Bb = W + col0 * Dc;
  v8f acc[2][2] = {};
  float4 ra[4], rb[2];
  load_f32<BM>(ra, Ab, Dc, tid);
  load_f32<BN>(rb, Bb, Dc, tid);
  for (int k0 = 0; k0 < Dc; k0 += BK) {
    __syncthreads();
    store_bf16<BM>(As, ra, tid);
    store_bf16<BN>(Bs, rb, tid);
    if (k0 + BK < Dc) {
      load_f32<BM>(ra, Ab + k0 + BK, Dc, tid);
      load_f32<BN>(rb, Bb + k0 + BK, Dc, tid);
      prefetch_rows<BM>(Ab + k0 + 2 * BK, Dc, tid);
      prefetch_rows<BN>(Bb + k0 + 2 * BK, Dc, tid);
    }
    __syncthreads();
    kblock_mma(As, Bs, wm, wn, lane, acc);
  }
  epilogue(acc, wm, wn, lane, row0, col0, [&](long m, long n, float cv) {
    const long b = m / Lc;
    const long l = m % Lc;
    const long h = n / HDc;
    const long hd = n % HDc;
    const long o = transpose_out ? (((b * Hc + h) * HDc + hd) * Lc + l)
                                 : (((b * Hc + h) * Lc + l) * HDc + hd);
    dst[o] = (__bf16)cv;
  });
}

// ---------------------------------------------------------------------------
// Kernel 2: energy  S = scale * Q @ K^T  per (b,h).  K=64 -> 2 K-steps, fully
// unrolled: all four tile DMAs issued up front, threshold waits in between.
// Per-thread async ops: A tile = 2, B tile = 1  (issue order A0 A0 B0 A1 A1 B1)
// ---------------------------------------------------------------------------
__global__ __launch_bounds__(NT) void energy_kernel(const __bf16* __restrict__ Qh,
                                                    const __bf16* __restrict__ Kh,
                                                    float* __restrict__ attn) {
  __shared__ alignas(16) __bf16 As0[BM * LDSS], As1[BM * LDSS];
  __shared__ alignas(16) __bf16 Bs0[BN * LDSS], Bs1[BN * LDSS];
  const int tid = threadIdx.x, lane = tid & 31, wid = tid >> 5;
  const int wm = wid & 3, wn = wid >> 2;
  const long bh = blockIdx.z;
  const __bf16* Aq = Qh + bh * (long)Lc * HDc;
  const __bf16* Bk = Kh + bh * (long)Lc * HDc;
  const long row0 = (long)blockIdx.x * BM;
  const long col0 = (long)blockIdx.y * BN;
  v8f acc[2][2] = {};
  stage_async_bf16<BM>(As0, Aq + row0 * HDc, HDc, tid);
  stage_async_bf16<BN>(Bs0, Bk + col0 * HDc, HDc, tid);
  stage_async_bf16<BM>(As1, Aq + row0 * HDc + BK, HDc, tid);
  stage_async_bf16<BN>(Bs1, Bk + col0 * HDc + BK, HDc, tid);
  wait_async3();        // first 3 ops (tile 0) retired
  __syncthreads();
  kblock_mma(As0, Bs0, wm, wn, lane, acc);
  wait_async0();        // tile 1 retired
  __syncthreads();
  kblock_mma(As1, Bs1, wm, wn, lane, acc);
  float* rowbase = attn + bh * (long)Lc * Lc;
  epilogue(acc, wm, wn, lane, row0, col0, [&](long m, long n, float cv) {
    rowbase[m * Lc + n] = cv * 0.125f;  // 1/sqrt(64)
  });
}

// ---------------------------------------------------------------------------
// Kernel 3: row softmax over attention [B*H*L rows, Lc cols], in place
// ---------------------------------------------------------------------------
__global__ __launch_bounds__(NT) void softmax_kernel(float* __restrict__ attn) {
  __shared__ float red[NT];
  float* rowp = attn + (long)blockIdx.x * Lc;
  const int tid = threadIdx.x;
  float v[Lc / NT];
  float mx = -INFINITY;
#pragma unroll
  for (int i = 0; i < Lc / NT; ++i) { v[i] = rowp[tid + i * NT]; mx = fmaxf(mx, v[i]); }
  red[tid] = mx; __syncthreads();
  for (int s = NT / 2; s > 0; s >>= 1) {
    if (tid < s) red[tid] = fmaxf(red[tid], red[tid + s]);
    __syncthreads();
  }
  mx = red[0]; __syncthreads();
  float sum = 0.f;
#pragma unroll
  for (int i = 0; i < Lc / NT; ++i) { v[i] = __expf(v[i] - mx); sum += v[i]; }
  red[tid] = sum; __syncthreads();
  for (int s = NT / 2; s > 0; s >>= 1) {
    if (tid < s) red[tid] += red[tid + s];
    __syncthreads();
  }
  const float inv = 1.0f / red[0];
#pragma unroll
  for (int i = 0; i < Lc / NT; ++i) rowp[tid + i * NT] = v[i] * inv;
}

// ---------------------------------------------------------------------------
// Kernel 4: ctx = P @ V  per (b,h)   (P f32 [L,L], Vt bf16 [HD,L] -> K-contig)
// A: register-pipelined cvt path (+prefetch). B: async DMA, ping-pong LDS,
// 1 async op/thread/tile -> s_wait_asynccnt 1 overlaps next tile's DMA.
// ---------------------------------------------------------------------------
__global__ __launch_bounds__(NT) void pv_kernel(const float* __restrict__ attn,
                                                const __bf16* __restrict__ Vt,
                                                __bf16* __restrict__ ctx) {
  __shared__ alignas(16) __bf16 As[BM * LDSS];
  __shared__ alignas(16) __bf16 Bs0[BN * LDSS], Bs1[BN * LDSS];
  const int tid = threadIdx.x, lane = tid & 31, wid = tid >> 5;
  const int wm = wid & 3, wn = wid >> 2;
  const long bh = blockIdx.z;
  const long row0 = (long)blockIdx.x * BM;
  const long col0 = 0;  // N = HD = 64 = BN, single N-block
  const float*  Ab = attn + bh * (long)Lc * Lc + row0 * Lc;
  const __bf16* Bb = Vt + bh * (long)HDc * Lc;
  v8f acc[2][2] = {};
  float4 ra[4];
  load_f32<BM>(ra, Ab, Lc, tid);
  stage_async_bf16<BN>(Bs0, Bb, Lc, tid);
  for (int k0 = 0; k0 < Lc; k0 += BK) {
    __bf16* Bcur = ((k0 / BK) & 1) ? Bs1 : Bs0;
    __bf16* Bnxt = ((k0 / BK) & 1) ? Bs0 : Bs1;
    __syncthreads();
    store_bf16<BM>(As, ra, tid);
    if (k0 + BK < Lc) {
      stage_async_bf16<BN>(Bnxt, Bb + k0 + BK, Lc, tid);
      load_f32<BM>(ra, Ab + k0 + BK, Lc, tid);
      prefetch_rows<BM>(Ab + k0 + 2 * BK, Lc, tid);
      wait_async1();      // current B tile retired; next tile still in flight
    } else {
      wait_async0();
    }
    __syncthreads();
    kblock_mma(As, Bcur, wm, wn, lane, acc);
  }
  const long b = bh / Hc, h = bh % Hc;
  epilogue(acc, wm, wn, lane, row0, col0, [&](long m, long n, float cv) {
    ctx[(b * Lc + m) * Dc + h * HDc + n] = (__bf16)cv;
  });
}

// ---------------------------------------------------------------------------
// Kernel 5: out = ctx @ Wo^T + bo
// A (ctx bf16): async DMA ping-pong, 2 ops/thread/tile -> s_wait_asynccnt 2.
// B (Wo f32): register-pipelined cvt path.
// ---------------------------------------------------------------------------
__global__ __launch_bounds__(NT) void outproj_kernel(const __bf16* __restrict__ ctx,
                                                     const float* __restrict__ Wo,
                                                     const float* __restrict__ bo,
                                                     float* __restrict__ out) {
  __shared__ alignas(16) __bf16 As0[BM * LDSS], As1[BM * LDSS];
  __shared__ alignas(16) __bf16 Bs[BN * LDSS];
  const int tid = threadIdx.x, lane = tid & 31, wid = tid >> 5;
  const int wm = wid & 3, wn = wid >> 2;
  const long row0 = (long)blockIdx.x * BM;
  const long col0 = (long)blockIdx.y * BN;
  const __bf16* Ab = ctx + row0 * Dc;
  const float*  Bb = Wo + col0 * Dc;
  v8f acc[2][2] = {};
  float4 rb[2];
  load_f32<BN>(rb, Bb, Dc, tid);
  stage_async_bf16<BM>(As0, Ab, Dc, tid);
  for (int k0 = 0; k0 < Dc; k0 += BK) {
    __bf16* Acur = ((k0 / BK) & 1) ? As1 : As0;
    __bf16* Anxt = ((k0 / BK) & 1) ? As0 : As1;
    __syncthreads();
    store_bf16<BN>(Bs, rb, tid);
    if (k0 + BK < Dc) {
      stage_async_bf16<BM>(Anxt, Ab + k0 + BK, Dc, tid);
      load_f32<BN>(rb, Bb + k0 + BK, Dc, tid);
      prefetch_rows<BN>(Bb + k0 + 2 * BK, Dc, tid);
      wait_async2();      // current A tile (2 ops) retired
    } else {
      wait_async0();
    }
    __syncthreads();
    kblock_mma(Acur, Bs, wm, wn, lane, acc);
  }
  epilogue(acc, wm, wn, lane, row0, col0, [&](long m, long n, float cv) {
    out[m * Dc + n] = cv + bo[n];
  });
}

// ---------------------------------------------------------------------------
extern "C" void kernel_launch(void* const* d_in, const int* in_sizes, int n_in,
                              void* d_out, int out_size, void* d_ws, size_t ws_size,
                              hipStream_t stream) {
  const float* query = (const float*)d_in[0];
  const float* key   = (const float*)d_in[1];
  const float* value = (const float*)d_in[2];
  const float* Wq    = (const float*)d_in[3];
  const float* Wk    = (const float*)d_in[4];
  const float* Wv    = (const float*)d_in[5];
  const float* Wo    = (const float*)d_in[6];
  const float* bo    = (const float*)d_in[7];

  float* out  = (float*)d_out;                       // [B,L,D]
  float* attn = out + (long)Bc * Lc * Dc;            // [B,H,L,L]

  __bf16* Qh  = (__bf16*)d_ws;                       // [B,H,L,HD]
  __bf16* Kh  = Qh + (long)Bc * Lc * Dc;             // [B,H,L,HD]
  __bf16* Vt  = Kh + (long)Bc * Lc * Dc;             // [B,H,HD,L]
  __bf16* ctx = Vt + (long)Bc * Lc * Dc;             // [B,L,D]

  const dim3 blk(NT);
  const dim3 gproj(Bc * Lc / BM, Dc / BN);           // 64 x 16
  proj_kernel<<<gproj, blk, 0, stream>>>(query, Wq, Qh, 0);
  proj_kernel<<<gproj, blk, 0, stream>>>(key,   Wk, Kh, 0);
  proj_kernel<<<gproj, blk, 0, stream>>>(value, Wv, Vt, 1);

  const dim3 ge(Lc / BM, Lc / BN, Bc * Hc);          // 16 x 32 x 64
  energy_kernel<<<ge, blk, 0, stream>>>(Qh, Kh, attn);

  softmax_kernel<<<dim3(Bc * Hc * Lc), blk, 0, stream>>>(attn);

  const dim3 gpv(Lc / BM, 1, Bc * Hc);               // 16 x 1 x 64
  pv_kernel<<<gpv, blk, 0, stream>>>(attn, Vt, ctx);

  const dim3 go(Bc * Lc / BM, Dc / BN);              // 64 x 16
  outproj_kernel<<<go, blk, 0, stream>>>(ctx, Wo, bo, out);
}
